// CrossAttn_4947802325504
// MI455X (gfx1250) — matrix operand order repaired and checked by reading, not compiled, not run on hardware
//
#include <hip/hip_runtime.h>

// Problem constants (reference: N=2, C=256, R=8 -> CK=32, T=4, H=32, W=32 -> L=4096)
#define NB 2
#define CCH 256
#define CKD 32
#define LL 4096
#define CHUNKS 8

typedef __attribute__((ext_vector_type(16))) __bf16 v16bf;
typedef __attribute__((ext_vector_type(8)))  float  v8f;

// ---------------------------------------------------------------- helpers ---

__device__ __forceinline__ __bf16 f2bf(float f) {
  unsigned u = __builtin_bit_cast(unsigned, f);
  unsigned r = u + 0x7FFFu + ((u >> 16) & 1u);   // round-to-nearest-even
  unsigned short h = (unsigned short)(r >> 16);
  return __builtin_bit_cast(__bf16, h);
}

__device__ __forceinline__ v8f zero8() {
  v8f z = {0.f, 0.f, 0.f, 0.f, 0.f, 0.f, 0.f, 0.f};
  return z;
}

__device__ __forceinline__ v8f wmma_bf16(v16bf a, v16bf b, v8f c) {
  // D = A(16x32 bf16) * B(32x16 bf16) + C(16x16 f32)
  return __builtin_amdgcn_wmma_f32_16x16x32_bf16(false, a, false, b,
                                                 (short)0, c, false, false);
}

// Low 32 bits of a generic pointer to __shared__ == wave-relative LDS offset
// (flat aperture rule: the LDS address is the low 32 bits of the address).
__device__ __forceinline__ unsigned lds_off(const void* p) {
  return (unsigned)(size_t)p;
}

// Async copy: 64 contiguous bytes global -> LDS (per lane), tracked by ASYNCcnt.
// IOFFSET is added to BOTH the global and the LDS address (ISA 10.x async ops),
// so a single base covers the whole 64B row.
__device__ __forceinline__ void async_copy64(unsigned dst, const void* src) {
  asm volatile(
      "global_load_async_to_lds_b128 %0, %1, off\n\t"
      "global_load_async_to_lds_b128 %0, %1, off offset:16\n\t"
      "global_load_async_to_lds_b128 %0, %1, off offset:32\n\t"
      "global_load_async_to_lds_b128 %0, %1, off offset:48"
      :: "v"(dst), "v"(src) : "memory");
}

// Async copy: 16 contiguous bytes global -> LDS (per lane).
__device__ __forceinline__ void async_copy16(unsigned dst, const void* src) {
  asm volatile("global_load_async_to_lds_b128 %0, %1, off"
               :: "v"(dst), "v"(src) : "memory");
}

__device__ __forceinline__ void wait_async_le4() {
  asm volatile("s_wait_asynccnt 0x4" ::: "memory");
}
__device__ __forceinline__ void wait_async_0() {
  asm volatile("s_wait_asynccnt 0x0" ::: "memory");
}

// A fragment: 16x32 bf16 tile, row-major in LDS with stride `ld`.
// Layout: lanes 0-15 hold row M=lane, K=0..15;
//         lanes 16-31 hold row M=lane-16, K=16..31.
__device__ __forceinline__ v16bf load_a_frag(const __bf16* t, int ld) {
  int lane = threadIdx.x & 31;
  int row  = lane & 15;
  int k0   = (lane >> 4) << 4;          // 0 or 16
  const __bf16* p = t + row * ld;
  v16bf a;
#pragma unroll
  for (int i = 0; i < 8; ++i) {
    a[i]     = p[k0 + i];
    a[8 + i] = p[k0 + 8 + i];
  }
  return a;
}

// B fragment: 32x16 bf16 tile stored row-major [K][N] with stride `ld`.
// Layout used here (contiguous k, same as load_a_frag): lanes 0-15 hold col N=lane, K=0..15; lanes 16-31 K=16..31.
__device__ __forceinline__ v16bf load_b_frag(const __bf16* t, int ld) {
  int lane = threadIdx.x & 31;
  int col  = lane & 15;
  int k0   = (lane >> 4) << 4;          // 0 or 16
  v16bf b;
#pragma unroll
  for (int i = 0; i < 16; ++i) b[i] = t[(k0 + i) * ld + col];
  return b;
}

// B fragment from a transposed tile: t is [N][K] row-major (stride ld).
// Each lane reads 16 contiguous bf16 -> conflict-friendly.
__device__ __forceinline__ v16bf load_b_fragT(const __bf16* t, int ld) {
  int lane = threadIdx.x & 31;
  int col  = lane & 15;
  int k0   = (lane >> 4) << 4;
  const __bf16* p = t + col * ld + k0;
  v16bf b;
#pragma unroll
  for (int i = 0; i < 16; ++i) b[i] = p[i];
  return b;
}

// ---------------------------------------------------- 1x1 conv projections ---
// out = W[rows x 256] * x[256 x L] + bias.  bf16 output.
// STORE_T=1: out stored [L x rows] (for k1/k2, so cor GEMM loads contiguously)
// STORE_T=0: out stored [rows x L] (for v1/v2, A-operand of the r GEMMs)
// (f32 -> bf16 conversion happens while staging, so this path stays synchronous.)
template <bool STORE_T>
__global__ __launch_bounds__(128) void proj_kernel(
    const float* __restrict__ W, const float* __restrict__ bias,
    const float* __restrict__ x, __bf16* __restrict__ out, int rows) {
  __shared__ __bf16 sA[16 * 40];
  __shared__ __bf16 sB[32 * 72];

  const int tid = threadIdx.x;
  const int lane = tid & 31, wave = tid >> 5;
  const int colbase = blockIdx.x * 64;
  const int rowbase = blockIdx.y * 16;
  const int n = blockIdx.z;

  const float* xn = x + (size_t)n * CCH * LL;
  __bf16* outn = out + (size_t)n * rows * LL;

  v8f acc = zero8();

  for (int kc = 0; kc < CCH; kc += 32) {
    // stage W tile 16x32 -> bf16
#pragma unroll
    for (int e = 0; e < 4; ++e) {
      int i = e * 128 + tid;            // 0..511
      int r = i >> 5, k = i & 31;
      sA[r * 40 + k] = f2bf(W[(size_t)(rowbase + r) * CCH + kc + k]);
    }
    // stage x tile 32x64 -> bf16 (coalesced along columns)
#pragma unroll
    for (int e = 0; e < 16; ++e) {
      int i = e * 128 + tid;            // 0..2047
      int cc = i & 63, kr = i >> 6;
      sB[kr * 72 + cc] = f2bf(xn[(size_t)(kc + kr) * LL + colbase + cc]);
    }
    __syncthreads();
    v16bf a = load_a_frag(sA, 40);
    v16bf b = load_b_frag(sB + wave * 16, 72);
    acc = wmma_bf16(a, b, acc);
    __syncthreads();
  }

  const int Mo = (lane >> 4) * 8;       // C/D layout: M = r + 8*(lane>>4)
  const int Nc = lane & 15;
#pragma unroll
  for (int r = 0; r < 8; ++r) {
    int row = rowbase + Mo + r;
    int col = colbase + wave * 16 + Nc;
    float val = acc[r] + bias[row];
    if (STORE_T)
      outn[(size_t)col * rows + row] = f2bf(val);
    else
      outn[(size_t)row * LL + col] = f2bf(val);
  }
}

// ------------------------------------------------------------- cor = k1^T k2 -
// k1t,k2t: [NB][LL][32] bf16.  cor: [NB][LL][LL] f32.
// 128x128 output tile per block; K=32 -> one WMMA per 16x16 tile.
// Tiles staged with async global->LDS copies (ASYNCcnt).
__global__ __launch_bounds__(256) void cor_kernel(
    const __bf16* __restrict__ k1t, const __bf16* __restrict__ k2t,
    float* __restrict__ cor) {
  __shared__ __bf16 sA[128 * 40];   // 80B rows (16B-aligned b128 LDS writes)
  __shared__ __bf16 sB[128 * 40];

  const int tid = threadIdx.x;
  const int lane = tid & 31, wave = tid >> 5;
  const int mbase = blockIdx.x * 128;
  const int lbase = blockIdx.y * 128;
  const int n = blockIdx.z;

  // both tiles are fully contiguous in global (row stride CKD=32 bf16 = 64B)
  const char* srcA = (const char*)(k1t + ((size_t)n * LL + lbase) * CKD);
  const char* srcB = (const char*)(k2t + ((size_t)n * LL + mbase) * CKD);
  // 128 rows x 64B per tile: 512 b128 chunks; 256 threads -> 2 per tile each
#pragma unroll
  for (int e = 0; e < 2; ++e) {
    int i = e * 256 + tid;              // chunk id 0..511
    int row = i >> 2, ch = (i & 3) * 16;
    async_copy16(lds_off(sA + row * 40) + ch, srcA + (size_t)row * 64 + ch);
    async_copy16(lds_off(sB + row * 40) + ch, srcB + (size_t)row * 64 + ch);
  }
  wait_async_0();
  __syncthreads();

  v16bf a = load_a_frag(sA + wave * 16 * 40, 40);
  float* corn = cor + (size_t)n * LL * LL;
  const int Mo = (lane >> 4) * 8;
  const int Nc = lane & 15;

#pragma unroll
  for (int ct = 0; ct < 8; ++ct) {
    v16bf b = load_b_fragT(sB + ct * 16 * 40, 40);
    v8f c = zero8();
    c = wmma_bf16(a, b, c);
#pragma unroll
    for (int r = 0; r < 8; ++r) {
      int l = lbase + wave * 16 + Mo + r;
      int m = mbase + ct * 16 + Nc;
      corn[(size_t)l * LL + m] = c[r];
    }
  }
}

// ---------------------------------------------------------- softmax statistics
__global__ __launch_bounds__(256) void rowstats_kernel(
    const float* __restrict__ cor, float* __restrict__ rmax,
    float* __restrict__ rsum) {
  __shared__ float red[256];
  const int tid = threadIdx.x;
  const int l = blockIdx.x, n = blockIdx.y;
  const float* row = cor + ((size_t)n * LL + l) * LL;

  float vals[16];
  float mx = -3.4e38f;
#pragma unroll
  for (int i = 0; i < 16; ++i) {
    vals[i] = row[i * 256 + tid];
    mx = fmaxf(mx, vals[i]);
  }
  red[tid] = mx;
  __syncthreads();
  for (int s = 128; s > 0; s >>= 1) {
    if (tid < s) red[tid] = fmaxf(red[tid], red[tid + s]);
    __syncthreads();
  }
  mx = red[0];
  __syncthreads();

  float sum = 0.f;
#pragma unroll
  for (int i = 0; i < 16; ++i) sum += __expf(vals[i] - mx);
  red[tid] = sum;
  __syncthreads();
  for (int s = 128; s > 0; s >>= 1) {
    if (tid < s) red[tid] += red[tid + s];
    __syncthreads();
  }
  if (tid == 0) {
    rmax[(size_t)n * LL + l] = mx;
    rsum[(size_t)n * LL + l] = red[0];
  }
}

__global__ __launch_bounds__(256) void colpart_kernel(
    const float* __restrict__ cor, float* __restrict__ pmax,
    float* __restrict__ psum) {
  const int tid = threadIdx.x;
  const int m = blockIdx.x * 256 + tid;
  const int chunk = blockIdx.y, n = blockIdx.z;
  const int l0 = chunk * (LL / CHUNKS);
  const float* base = cor + (size_t)n * LL * LL;

  float M = -3.4e38f, S = 0.f;
  for (int l = l0; l < l0 + LL / CHUNKS; ++l) {
    float v = base[(size_t)l * LL + m];            // coalesced across threads
    if (v > M) {
      S = S * __expf(M - v) + 1.0f;
      M = v;
    } else {
      S += __expf(v - M);
    }
  }
  size_t idx = ((size_t)n * CHUNKS + chunk) * LL + m;
  pmax[idx] = M;
  psum[idx] = S;
}

__global__ __launch_bounds__(256) void colfinal_kernel(
    const float* __restrict__ pmax, const float* __restrict__ psum,
    float* __restrict__ cmax, float* __restrict__ csum) {
  const int tid = threadIdx.x;
  const int m = blockIdx.x * 256 + tid;
  const int n = blockIdx.y;
  float M = -3.4e38f, S = 0.f;
#pragma unroll
  for (int c = 0; c < CHUNKS; ++c) {
    size_t idx = ((size_t)n * CHUNKS + c) * LL + m;
    float pm = pmax[idx], ps = psum[idx];
    if (pm > M) {
      S = S * __expf(M - pm) + ps;
      M = pm;
    } else {
      S += ps * __expf(pm - M);
    }
  }
  cmax[(size_t)n * LL + m] = M;
  csum[(size_t)n * LL + m] = S;
}

// ---------------------------------------------- r GEMM + softmax + residual ---
// TRANSP=0: out1[c,m] = x1[c,m] + sum_l v1[c,l] * exp(cor[l,m]-rmax[l])/rsum[l]
// TRANSP=1: out2[c,l] = x2[c,l] + sum_m v2[c,m] * exp(cor[l,m]-cmax[m])/csum[m]
// Block: 256 threads / 8 waves; output tile 256(C) x 64 cols; K over L in 32s.
// A tile (v) is double-buffered via async global->LDS; the prefetch for chunk
// i+1 overlaps the exp/cvt B-tile build and the 8 WMMAs of chunk i.  Async
// loads complete in order, so s_wait_asynccnt<=4 retires exactly the current
// chunk while the 4 prefetch ops stay in flight.
template <bool TRANSP>
__global__ __launch_bounds__(256) void attn_kernel(
    const __bf16* __restrict__ v, const float* __restrict__ cor,
    const float* __restrict__ smax, const float* __restrict__ ssum,
    const float* __restrict__ x, float* __restrict__ out) {
  __shared__ __bf16 sA[2][256 * 40];  // v tile [256 x 32], 80B rows, dbl-buffered
  __shared__ __bf16 sB[32 * 72];      // attn tile [32 x 64]

  const int tid = threadIdx.x;
  const int lane = tid & 31, wave = tid >> 5;
  const int colbase = blockIdx.x * 64;
  const int n = blockIdx.y;

  const __bf16* vn = v + (size_t)n * CCH * LL;
  const float* corn = cor + (size_t)n * LL * LL;
  const float* mxp = smax + (size_t)n * LL;
  const float* smp = ssum + (size_t)n * LL;

  const char* vrow = (const char*)(vn + (size_t)tid * LL);  // this thread's row
  const unsigned dstA0 = lds_off(&sA[0][tid * 40]);
  const unsigned dstA1 = lds_off(&sA[1][tid * 40]);

  v8f acc[2][4];
#pragma unroll
  for (int i = 0; i < 2; ++i)
#pragma unroll
    for (int j = 0; j < 4; ++j) acc[i][j] = zero8();

  const int NIT = LL / 32;
  // prologue: chunk 0 -> buffer 0
  async_copy64(dstA0, vrow);

  for (int it = 0; it < NIT; ++it) {
    const int kk = it * 32;
    const int p = it & 1;

    // prefetch chunk it+1 into the other buffer (its readers finished at the
    // trailing barrier of iteration it-1)
    if (it + 1 < NIT)
      async_copy64(p ? dstA0 : dstA1, vrow + (size_t)(kk + 32) * 2);

    // stage B: normalized attention probabilities (bf16) -- VALU transform
#pragma unroll
    for (int e = 0; e < 8; ++e) {
      int i = e * 256 + tid;
      int kr, cc;
      float cv;
      if (TRANSP) {
        cc = i >> 5;
        kr = i & 31;                   // 32 consecutive floats per row -> coalesced
        cv = corn[(size_t)(colbase + cc) * LL + kk + kr];
      } else {
        cc = i & 63;
        kr = i >> 6;                   // 64 consecutive floats per row -> coalesced
        cv = corn[(size_t)(kk + kr) * LL + colbase + cc];
      }
      float p2 = __expf(cv - mxp[kk + kr]) * (1.0f / smp[kk + kr]);
      sB[kr * 72 + cc] = f2bf(p2);
    }

    // retire this chunk's 4 async ops (prefetch may remain outstanding)
    if (it + 1 < NIT) wait_async_le4();
    else              wait_async_0();
    __syncthreads();

    v16bf bfr[4];
#pragma unroll
    for (int tj = 0; tj < 4; ++tj) bfr[tj] = load_b_frag(sB + tj * 16, 72);
#pragma unroll
    for (int ti = 0; ti < 2; ++ti) {
      v16bf a = load_a_frag(&sA[p][(wave * 32 + ti * 16) * 40], 40);
#pragma unroll
      for (int tj = 0; tj < 4; ++tj)
        acc[ti][tj] = wmma_bf16(a, bfr[tj], acc[ti][tj]);
    }
    __syncthreads();
  }

  const int Mo = (lane >> 4) * 8;
  const int Nc = lane & 15;
#pragma unroll
  for (int ti = 0; ti < 2; ++ti)
#pragma unroll
    for (int tj = 0; tj < 4; ++tj)
#pragma unroll
      for (int r = 0; r < 8; ++r) {
        int c = wave * 32 + ti * 16 + Mo + r;
        int col = colbase + tj * 16 + Nc;
        size_t idx = ((size_t)n * CCH + c) * LL + col;
        out[idx] = x[idx] + acc[ti][tj][r];
      }
}

// ------------------------------------------------------------------- launch --

extern "C" void kernel_launch(void* const* d_in, const int* in_sizes, int n_in,
                              void* d_out, int out_size, void* d_ws,
                              size_t ws_size, hipStream_t stream) {
  const float* x1  = (const float*)d_in[0];
  const float* x2  = (const float*)d_in[1];
  const float* Wk1 = (const float*)d_in[2];
  const float* bk1 = (const float*)d_in[3];
  const float* Wk2 = (const float*)d_in[4];
  const float* bk2 = (const float*)d_in[5];
  const float* Wv1 = (const float*)d_in[6];
  const float* bv1 = (const float*)d_in[7];
  const float* Wv2 = (const float*)d_in[8];
  const float* bv2 = (const float*)d_in[9];

  float* out1 = (float*)d_out;
  float* out2 = out1 + (size_t)NB * CCH * LL;

  // Workspace carve-out (~137.5 MB total; cor dominates at 128 MB, L2-resident)
  char* ws = (char*)d_ws;
  size_t off = 0;
  auto carve = [&](size_t bytes) -> void* {
    void* p = ws + off;
    off += (bytes + 255) & ~(size_t)255;
    return p;
  };
  __bf16* k1t = (__bf16*)carve((size_t)NB * LL * CKD * 2);   // [NB][L][32]
  __bf16* k2t = (__bf16*)carve((size_t)NB * LL * CKD * 2);
  __bf16* v1b = (__bf16*)carve((size_t)NB * CCH * LL * 2);   // [NB][C][L]
  __bf16* v2b = (__bf16*)carve((size_t)NB * CCH * LL * 2);
  float*  cor = (float*) carve((size_t)NB * LL * LL * 4);    // [NB][L][L]
  float* rmax = (float*) carve((size_t)NB * LL * 4);
  float* rsum = (float*) carve((size_t)NB * LL * 4);
  float* cmax = (float*) carve((size_t)NB * LL * 4);
  float* csum = (float*) carve((size_t)NB * LL * 4);
  float* pmax = (float*) carve((size_t)NB * CHUNKS * LL * 4);
  float* psum = (float*) carve((size_t)NB * CHUNKS * LL * 4);
  (void)ws_size; (void)in_sizes; (void)n_in; (void)out_size;

  // 1) projections (bf16 outputs; k's transposed for contiguous cor operands)
  proj_kernel<true ><<<dim3(LL / 64, CKD / 16, NB), 128, 0, stream>>>(Wk1, bk1, x1, k1t, CKD);
  proj_kernel<true ><<<dim3(LL / 64, CKD / 16, NB), 128, 0, stream>>>(Wk2, bk2, x2, k2t, CKD);
  proj_kernel<false><<<dim3(LL / 64, CCH / 16, NB), 128, 0, stream>>>(Wv1, bv1, x1, v1b, CCH);
  proj_kernel<false><<<dim3(LL / 64, CCH / 16, NB), 128, 0, stream>>>(Wv2, bv2, x2, v2b, CCH);

  // 2) correlation matrix (f32, K=32 -> single WMMA per tile)
  cor_kernel<<<dim3(LL / 128, LL / 128, NB), 256, 0, stream>>>(k1t, k2t, cor);

  // 3) dual softmax statistics
  rowstats_kernel<<<dim3(LL, NB), 256, 0, stream>>>(cor, rmax, rsum);
  colpart_kernel<<<dim3(LL / 256, CHUNKS, NB), 256, 0, stream>>>(cor, pmax, psum);
  colfinal_kernel<<<dim3(LL / 256, NB), 256, 0, stream>>>(pmax, psum, cmax, csum);

  // 4) attention GEMMs with on-the-fly softmax + residual
  attn_kernel<false><<<dim3(LL / 64, NB), 256, 0, stream>>>(v1b, cor, rmax, rsum, x1, out1);
  attn_kernel<true ><<<dim3(LL / 64, NB), 256, 0, stream>>>(v2b, cor, cmax, csum, x2, out2);
}
